// HyperParamNet_34557306864278
// MI455X (gfx1250) — matrix operand order repaired and checked by reading, not compiled, not run on hardware
//
#include <hip/hip_runtime.h>
#include <hip/hip_bf16.h>

typedef __attribute__((ext_vector_type(16))) _Float16 v16h;
typedef __attribute__((ext_vector_type(8)))  float    v8f;

union V8F  { v8f  v; float    f[8];  };
union V16H { v16h v; _Float16 h[16]; };

// ---------------------------------------------------------------------------
// Workspace layout (bytes):
//   [0      , 16384  )  wsA  : fw2 WMMA A-fragments, 16 frags x 32 lanes x 16 f16
//   [16384  , 21504  )  wsA2 : bw2 im2col A-fragments, 5 frags x 32 x 16 f16
//   [32768  , 557056 )  psum : alpha sum partials, 8 batch x 512 blk x 32 f32
//   [557056 , 1081344)  pmax : alpha max partials, 8 batch x 512 blk x 32 f32
//   [1081344, +4MB   )  xc   : (8,2,256,256) f32  [ch-max, ch-mean of tb]
// ---------------------------------------------------------------------------

// A-fragment half index -> K within a 32-wide K-step (contiguous, matches B):
//   lane<16 : h in 0..15 -> K=h       (0..15)
//   lane>=16: h in 0..15 -> K=h+16    (16..31)
__device__ __forceinline__ int afrag_k(int h, int hi) {
    return h + (hi ? 16 : 0);
}

__global__ __launch_bounds__(512) void prep_kernel(
    const float* __restrict__ fw2, const float* __restrict__ bw2,
    _Float16* __restrict__ wsA, _Float16* __restrict__ wsA2)
{
    const int tid  = threadIdx.x;
    const int lane = tid & 31;
    const int hi   = (lane >> 4) & 1;
    const int lrow = lane & 15;

    // fw2 (64 x 112) -> 4 M-tiles x 4 K-steps (K padded to 128)
    {
        const int frag = tid >> 5;            // 0..15
        const int mt = frag >> 2, kt = frag & 3;
        const int M  = mt * 16 + lrow;
#pragma unroll
        for (int h = 0; h < 16; ++h) {
            const int Kg = kt * 32 + afrag_k(h, hi);
            wsA[((size_t)frag * 32 + lane) * 16 + h] =
                (Kg < 112) ? (_Float16)fw2[M * 112 + Kg] : (_Float16)0.f;
        }
    }
    // bw2 (16 x 144 im2col, u = ic*9+kh*3+kw) -> 1 M-tile x 5 K-steps (pad 160)
    if (tid < 160) {
        const int frag = tid >> 5;            // 0..4
        const int M = lrow;
#pragma unroll
        for (int h = 0; h < 16; ++h) {
            const int Kg = frag * 32 + afrag_k(h, hi);
            wsA2[((size_t)frag * 32 + lane) * 16 + h] =
                (Kg < 144) ? (_Float16)bw2[M * 144 + Kg] : (_Float16)0.f;
        }
    }
}

// ---------------------------------------------------------------------------
// Kernel 1: temp -> grouped conv3x3(28 groups)+ReLU -> 1x1 conv (WMMA) -> t
// fused reduction epilogue (alpha partials + xc), t never hits memory.
// Block = 256 threads = 8 waves, one 8x16 spatial tile; wave w owns tile
// row w (one 16-pixel WMMA column block) -> only 4 accumulator tiles live.
// Interior tiles stage the input chunk with global_load_async_to_lds_b32
// (ASYNCcnt path, no VGPR round-trip); border tiles use the guarded path.
// ---------------------------------------------------------------------------
__global__ __launch_bounds__(256, 4) void fused_front_kernel(
    const float* __restrict__ y, const float* __restrict__ mask,
    const float* __restrict__ fw1, const float* __restrict__ fb1,
    const float* __restrict__ fb2,
    const _Float16* __restrict__ wsA,
    float* __restrict__ psum, float* __restrict__ pmax,
    float* __restrict__ xcbuf)
{
    __shared__ float s_chunk[16][10][18];   // 16 temp-channels, 8 rows + halo
    __shared__ float s_w1[32][18];          // fw1 slice for current K-step
    __shared__ float s_fb[32];
    __shared__ float s_fb2[64];
    __shared__ float s_red_sum[16][16];
    __shared__ float s_red_max[16][16];

    const int tid  = threadIdx.x;
    const int lane = tid & 31;
    const int w    = tid >> 5;        // wave 0..7 == tile row
    const int hi   = (lane >> 4) & 1;
    const int lp   = lane & 15;       // pixel column / WMMA N
    const int tx0 = blockIdx.x * 16, ty0 = blockIdx.y * 8;
    const int b   = blockIdx.z;

    // uniform per block: no zero padding needed anywhere in the halo
    const bool interior = (blockIdx.x > 0) && (blockIdx.x < 15) &&
                          (blockIdx.y > 0) && (blockIdx.y < 31);
    // flat LDS aperture address: low 32 bits == LDS byte offset
    const unsigned lds_base = (unsigned)(uintptr_t)&s_chunk[0][0][0];

    if (tid < 64) s_fb2[tid] = fb2[tid];

    V8F acc[4] = {};

    for (int kt = 0; kt < 4; ++kt) {
        __syncthreads();
        // temp channels kt*16 .. kt*16+15 (even = y, odd = mask)
        if (interior) {
            for (int i = tid; i < 16 * 10 * 18; i += 256) {
                const int t = i / 180, rem = i % 180, rr = rem / 18, cc = rem % 18;
                const int chg = kt * 16 + t;
                const unsigned lds = lds_base + (unsigned)i * 4u;
                if (chg < 56) {
                    const float* src = (chg & 1) ? mask : y;
                    const float* gp = src +
                        (((size_t)b * 28 + (chg >> 1)) * 256 + (ty0 - 1 + rr)) * 256 +
                        (tx0 - 1 + cc);
                    asm volatile("global_load_async_to_lds_b32 %0, %1, off"
                                 :: "v"(lds), "v"((unsigned long long)(uintptr_t)gp)
                                 : "memory");
                } else {
                    ((float*)s_chunk)[i] = 0.f;
                }
            }
        } else {
            for (int i = tid; i < 16 * 10 * 18; i += 256) {
                const int t = i / 180, rem = i % 180, rr = rem / 18, cc = rem % 18;
                const int chg = kt * 16 + t;
                const int gy = ty0 - 1 + rr, gx = tx0 - 1 + cc;
                float v = 0.f;
                if (chg < 56 && gy >= 0 && gy < 256 && gx >= 0 && gx < 256) {
                    const float* src = (chg & 1) ? mask : y;
                    v = src[(((size_t)b * 28 + (chg >> 1)) * 256 + gy) * 256 + gx];
                }
                ((float*)s_chunk)[i] = v;
            }
        }
        // fw1 slice: output channels kt*32 .. kt*32+31 (zero beyond 112)
        for (int i = tid; i < 32 * 18; i += 256) {
            const int o = i / 18, j = i % 18, ocg = kt * 32 + o;
            ((float*)s_w1)[i] = (ocg < 112) ? fw1[ocg * 18 + j] : 0.f;
        }
        if (tid < 32) {
            const int ocg = kt * 32 + tid;
            s_fb[tid] = (ocg < 112) ? fb1[ocg] : 0.f;
        }
        if (interior)
            asm volatile("s_wait_asynccnt 0x0" ::: "memory");
        __syncthreads();

        // A-fragments for this K-step
        V16H afr[4];
#pragma unroll
        for (int mt = 0; mt < 4; ++mt)
            afr[mt].v = *(const v16h*)(wsA + ((size_t)(mt * 4 + kt) * 32 + lane) * 16);

        {
            const int r0 = w;             // tile row; chunk row r0..r0+2
            const int c0 = lp;            // tile col; chunk col c0..c0+2
            V16H bf;
            // each lane computes 16 conv1+ReLU outputs = its B-frag K slice
#pragma unroll
            for (int gr = 0; gr < 4; ++gr) {
                const int gl = hi * 4 + gr;       // local group 0..7
                float tap[18];
#pragma unroll
                for (int ci = 0; ci < 2; ++ci)
#pragma unroll
                    for (int kh = 0; kh < 3; ++kh)
#pragma unroll
                        for (int kw = 0; kw < 3; ++kw)
                            tap[ci * 9 + kh * 3 + kw] =
                                s_chunk[gl * 2 + ci][r0 + kh][c0 + kw];
#pragma unroll
                for (int j = 0; j < 4; ++j) {
                    const int ol = gl * 4 + j;    // local oc 0..31
                    float a = s_fb[ol];
#pragma unroll
                    for (int k = 0; k < 18; ++k) a += s_w1[ol][k] * tap[k];
                    bf.h[gr * 4 + j] = (_Float16)fmaxf(a, 0.f);
                }
            }
#pragma unroll
            for (int mt = 0; mt < 4; ++mt) {
                acc[mt].v = __builtin_amdgcn_wmma_f32_16x16x32_f16(
                    false, afr[mt].v, false, bf.v, (short)0, acc[mt].v,
                    false, false);
            }
        }
    }

    // ---- epilogue: D VGPR r holds (M = r + hi*8, N = lp) => ch = mt*16+hi*8+r
    float asum[16], amax[16];
    {
        float tbmax = -3.4e38f, tbsum = 0.f;
#pragma unroll
        for (int mt = 0; mt < 4; ++mt) {
#pragma unroll
            for (int r = 0; r < 8; ++r) {
                const int ch = mt * 16 + hi * 8 + r;
                const float v = acc[mt].f[r] + s_fb2[ch];
                if (mt < 2) {                       // ta -> alpha reductions
                    asum[mt * 8 + r] = v;
                    amax[mt * 8 + r] = v;
                } else {                            // tb -> xc per pixel
                    tbmax = fmaxf(tbmax, v);
                    tbsum += v;
                }
            }
        }
        // combine the two half-wave channel sets of this pixel
        tbmax = fmaxf(tbmax, __shfl_xor(tbmax, 16, 32));
        tbsum += __shfl_xor(tbsum, 16, 32);
        if (lane < 16) {
            const int gy = ty0 + w, gx = tx0 + lp;
            xcbuf[(((size_t)b * 2 + 0) * 256 + gy) * 256 + gx] = tbmax;
            xcbuf[(((size_t)b * 2 + 1) * 256 + gy) * 256 + gx] = tbsum * (1.f / 32.f);
        }
    }

    // butterfly across the 16 pixels of each half-wave (deterministic)
#pragma unroll
    for (int i = 0; i < 16; ++i) {
#pragma unroll
        for (int m = 1; m <= 8; m <<= 1) {
            asum[i] += __shfl_xor(asum[i], m, 32);
            amax[i] = fmaxf(amax[i], __shfl_xor(amax[i], m, 32));
        }
    }
    if (lp == 0) {
#pragma unroll
        for (int i = 0; i < 16; ++i) {
            s_red_sum[w * 2 + hi][i] = asum[i];
            s_red_max[w * 2 + hi][i] = amax[i];
        }
    }
    __syncthreads();
    if (tid < 32) {                    // j = hi*16 + (mt*8+r)
        const int jhi = tid >> 4, ji = tid & 15;
        float s = 0.f, m = -3.4e38f;
#pragma unroll
        for (int ww = 0; ww < 8; ++ww) {
            s += s_red_sum[ww * 2 + jhi][ji];
            m = fmaxf(m, s_red_max[ww * 2 + jhi][ji]);
        }
        const int blk = blockIdx.y * 16 + blockIdx.x;     // 0..511
        psum[((size_t)b * 512 + blk) * 32 + tid] = s;
        pmax[((size_t)b * 512 + blk) * 32 + tid] = m;
    }
}

// ---------------------------------------------------------------------------
// Kernel 2: finish alpha = pgm(avg) + pgm(max); 8 blocks x 32 threads
// ---------------------------------------------------------------------------
__global__ __launch_bounds__(32) void alpha_kernel(
    const float* __restrict__ psum, const float* __restrict__ pmax,
    const float* __restrict__ aw1, const float* __restrict__ ab1,
    const float* __restrict__ aw2, const float* __restrict__ ab2,
    const float* __restrict__ aw3, const float* __restrict__ ab3,
    float* __restrict__ out_alpha)
{
    const int b = blockIdx.x, t = threadIdx.x;   // t = channel 0..31
    __shared__ float s0[32], s1[32];

    // channel -> partial slot j = hi*16 + (mt*8+r)
    const int hi = (t >> 3) & 1;
    const int j  = hi * 16 + (t >> 4) * 8 + (t & 7);

    float s = 0.f, m = -3.4e38f;
    for (int blk = 0; blk < 512; ++blk) {
        s += psum[((size_t)b * 512 + blk) * 32 + j];
        m = fmaxf(m, pmax[((size_t)b * 512 + blk) * 32 + j]);
    }
    const float x_avg = s * (1.f / 65536.f);
    const float x_max = m;

    float alpha_acc = 0.f;
    for (int src = 0; src < 2; ++src) {
        s0[t] = (src == 0) ? x_avg : x_max;
        __syncthreads();
        float h1 = ab1[t];
        for (int k = 0; k < 32; ++k) h1 += aw1[t * 32 + k] * s0[k];
        h1 = (h1 >= 0.f) ? h1 : 0.01f * h1;
        s1[t] = h1;
        __syncthreads();
        float h2 = ab2[t];
        for (int k = 0; k < 32; ++k) h2 += aw2[t * 32 + k] * s1[k];
        h2 = (h2 >= 0.f) ? h2 : 0.01f * h2;
        s0[t] = h2;
        __syncthreads();
        float o = ab3[0];
        for (int k = 0; k < 32; ++k) o += aw3[k] * s0[k];
        alpha_acc += fmaxf(o, 0.f) + 1e-6f;
        __syncthreads();
    }
    if (t == 0) out_alpha[b] = alpha_acc;
}

// ---------------------------------------------------------------------------
// Kernel 3: beta = sigmoid(conv3(leaky(conv2(leaky(conv1(xc))))))
// fused 16x16 tile with halo chain 22->20->18->16; conv2 (16->16,K=144) = WMMA
// ---------------------------------------------------------------------------
__global__ __launch_bounds__(256, 4) void beta_kernel(
    const float* __restrict__ xcbuf,
    const float* __restrict__ bw1, const float* __restrict__ bb1,
    const float* __restrict__ bb2,
    const float* __restrict__ bw3, const float* __restrict__ bb3,
    const _Float16* __restrict__ wsA2,
    float* __restrict__ out_beta)
{
    __shared__ float    s_xc[2][22][22];
    __shared__ _Float16 s_b1[16][20][20];
    __shared__ float    s_b2[16][18][18];
    __shared__ float    s_w1[16][18];
    __shared__ float    s_w3[144];
    __shared__ float    s_bb1[16], s_bb2[16];

    const int tid  = threadIdx.x;
    const int lane = tid & 31;
    const int w    = tid >> 5;
    const int hi   = (lane >> 4) & 1;
    const int lp   = lane & 15;
    const int tx0 = blockIdx.x * 16, ty0 = blockIdx.y * 16;
    const int b   = blockIdx.z;

    for (int i = tid; i < 2 * 22 * 22; i += 256) {
        const int c = i / 484, rem = i % 484, rr = rem / 22, cc = rem % 22;
        const int gy = ty0 - 3 + rr, gx = tx0 - 3 + cc;
        float v = 0.f;
        if (gy >= 0 && gy < 256 && gx >= 0 && gx < 256)
            v = xcbuf[(((size_t)b * 2 + c) * 256 + gy) * 256 + gx];
        ((float*)s_xc)[i] = v;
    }
    for (int i = tid; i < 288; i += 256) ((float*)s_w1)[i] = bw1[i];
    if (tid < 144) s_w3[tid] = bw3[tid];
    if (tid < 16) { s_bb1[tid] = bb1[tid]; s_bb2[tid] = bb2[tid]; }
    __syncthreads();

    // conv1 (2->16, leaky) on 20x20 grid; zero outside image (pad semantics)
    for (int p = tid; p < 400; p += 256) {
        const int r = p / 20, c = p % 20;
        const int gy = ty0 - 2 + r, gx = tx0 - 2 + c;
        const bool inimg = (gy >= 0 && gy < 256 && gx >= 0 && gx < 256);
#pragma unroll
        for (int oc = 0; oc < 16; ++oc) {
            float a = s_bb1[oc];
#pragma unroll
            for (int ci = 0; ci < 2; ++ci)
#pragma unroll
                for (int kh = 0; kh < 3; ++kh)
#pragma unroll
                    for (int kw = 0; kw < 3; ++kw)
                        a += s_w1[oc][ci * 9 + kh * 3 + kw] *
                             s_xc[ci][r + kh][c + kw];
            a = (a >= 0.f) ? a : 0.01f * a;
            s_b1[oc][r][c] = inimg ? (_Float16)a : (_Float16)0.f;
        }
    }
    __syncthreads();

    // conv2 (16->16, K=144 im2col, leaky) on 18x18 grid via WMMA
    for (int chunk = w; chunk < 21; chunk += 8) {   // uniform per wave
        const int p = chunk * 16 + lp;
        const int valid = (p < 324);
        const int pr = valid ? p / 18 : 0;
        const int pc = valid ? p % 18 : 0;
        V8F acc = {};
#pragma unroll
        for (int kt = 0; kt < 5; ++kt) {
            V16H bf;
#pragma unroll
            for (int h = 0; h < 16; ++h) {
                const int u = kt * 32 + hi * 16 + h;   // im2col index
                _Float16 v = (_Float16)0.f;
                if (u < 144) {
                    const int ic = u / 9, tt = u % 9;
                    v = s_b1[ic][pr + tt / 3][pc + tt % 3];
                }
                bf.h[h] = v;
            }
            const v16h av = *(const v16h*)(wsA2 + ((size_t)kt * 32 + lane) * 16);
            acc.v = __builtin_amdgcn_wmma_f32_16x16x32_f16(
                false, av, false, bf.v, (short)0, acc.v, false, false);
        }
        const int gy = ty0 - 1 + pr, gx = tx0 - 1 + pc;
        const bool inimg = (gy >= 0 && gy < 256 && gx >= 0 && gx < 256);
#pragma unroll
        for (int r = 0; r < 8; ++r) {
            const int ch = hi * 8 + r;
            float v2 = acc.f[r] + s_bb2[ch];
            v2 = (v2 >= 0.f) ? v2 : 0.01f * v2;
            if (valid) s_b2[ch][pr][pc] = inimg ? v2 : 0.f;
        }
    }
    __syncthreads();

    // conv3 (16->1) + sigmoid, one output pixel per thread
    {
        const int r = tid / 16, c = tid % 16;
        float a = bb3[0];
#pragma unroll
        for (int ic = 0; ic < 16; ++ic)
#pragma unroll
            for (int kh = 0; kh < 3; ++kh)
#pragma unroll
                for (int kw = 0; kw < 3; ++kw)
                    a += s_w3[ic * 9 + kh * 3 + kw] *
                         s_b2[ic][r + kh][c + kw];
        const float beta = 1.f / (1.f + expf(-a));
        out_beta[((size_t)b * 256 + (ty0 + r)) * 256 + (tx0 + c)] = beta;
    }
}

// ---------------------------------------------------------------------------
extern "C" void kernel_launch(void* const* d_in, const int* in_sizes, int n_in,
                              void* d_out, int out_size, void* d_ws, size_t ws_size,
                              hipStream_t stream) {
    const float* y    = (const float*)d_in[0];
    const float* mask = (const float*)d_in[1];
    const float* fw1  = (const float*)d_in[2];
    const float* fb1  = (const float*)d_in[3];
    const float* fw2  = (const float*)d_in[4];
    const float* fb2  = (const float*)d_in[5];
    const float* aw1  = (const float*)d_in[6];
    const float* ab1  = (const float*)d_in[7];
    const float* aw2  = (const float*)d_in[8];
    const float* ab2  = (const float*)d_in[9];
    const float* aw3  = (const float*)d_in[10];
    const float* ab3  = (const float*)d_in[11];
    const float* bw1  = (const float*)d_in[12];
    const float* bb1  = (const float*)d_in[13];
    const float* bw2  = (const float*)d_in[14];
    const float* bb2  = (const float*)d_in[15];
    const float* bw3  = (const float*)d_in[16];
    const float* bb3  = (const float*)d_in[17];
    (void)in_sizes; (void)n_in; (void)out_size; (void)ws_size;

    char* ws = (char*)d_ws;
    _Float16* wsA  = (_Float16*)(ws);
    _Float16* wsA2 = (_Float16*)(ws + 16384);
    float* psum    = (float*)(ws + 32768);
    float* pmax    = (float*)(ws + 557056);
    float* xcbuf   = (float*)(ws + 1081344);

    float* out = (float*)d_out;

    prep_kernel<<<1, 512, 0, stream>>>(fw2, bw2, wsA, wsA2);
    fused_front_kernel<<<dim3(16, 32, 8), 256, 0, stream>>>(
        y, mask, fw1, fb1, fb2, wsA, psum, pmax, xcbuf);
    alpha_kernel<<<8, 32, 0, stream>>>(
        psum, pmax, aw1, ab1, aw2, ab2, aw3, ab3, out);
    beta_kernel<<<dim3(16, 16, 8), 256, 0, stream>>>(
        xcbuf, bw1, bb1, bb2, bw3, bb3, wsA2, out + 8);
}
